// GroupedQueryAttention_65317862638159
// MI455X (gfx1250) — matrix operand order repaired and checked by reading, not compiled, not run on hardware
//
#include <hip/hip_runtime.h>
#include <hip/hip_bf16.h>
#include <math.h>

// Problem constants (from reference)
#define BB   2
#define TT   2048
#define CC   2048
#define HH   16
#define HKV  4
#define DD   128
// M = B*T = 4096 rows for all GEMMs

typedef __attribute__((ext_vector_type(16))) __bf16 v16bf;
typedef __attribute__((ext_vector_type(8)))  __bf16 v8bf;
typedef __attribute__((ext_vector_type(4)))  __bf16 v4bf;
typedef __attribute__((ext_vector_type(8)))  float  v8f;
typedef __attribute__((ext_vector_type(4)))  float  v4f;

#if __has_builtin(__builtin_amdgcn_sched_group_barrier)
#define SCHED_GROUP(mask, size, id) __builtin_amdgcn_sched_group_barrier(mask, size, id)
#else
#define SCHED_GROUP(mask, size, id)
#endif
// sched group masks: 0x008 = MFMA/WMMA, 0x020 = VMEM read

static __device__ __forceinline__ v8f wmma_bf16(v16bf a, v16bf b, v8f c) {
  // D = A(16x32 bf16) * B(32x16 bf16) + C(16x16 f32)
  return __builtin_amdgcn_wmma_f32_16x16x32_bf16(false, a, false, b, (short)0, c,
                                                 false, false);
}

// A-fragment: lane holds row M=lane%16, K elems {base..base+15} (contiguous k order, same as B),
// base = (lane/16)*16  -> two contiguous 8x bf16 (16B) loads.
static __device__ __forceinline__ v16bf pack_a(const __bf16* p0, const __bf16* p1) {
  v8bf lo = *(const v8bf*)p0;
  v8bf hi = *(const v8bf*)p1;
  v16bf r;
#pragma unroll
  for (int i = 0; i < 8; ++i) { r[i] = lo[i]; r[i + 8] = hi[i]; }
  return r;
}

// Load one k-step of fragments: A (16x32) + 8 B frags (32x16 each).
// 18 VMEM reads total (2 for A, 2 per B fragment).
static __device__ __forceinline__ void load_frags(const __bf16* __restrict__ xrow,
                                                  const __bf16* __restrict__ wrow,
                                                  int k0, int half,
                                                  v16bf& a, v16bf (&b)[8]) {
  const __bf16* ap = xrow + k0 + half * 16;
  a = pack_a(ap, ap + 8);
#pragma unroll
  for (int ct = 0; ct < 8; ++ct) {
    const __bf16* bp = wrow + (size_t)ct * 16 * CC + k0 + half * 16;
    b[ct] = *(const v16bf*)bp;
  }
}

// ---------------------------------------------------------------- converts
__global__ __launch_bounds__(256)
void cvt_f32_bf16(const float* __restrict__ src, __bf16* __restrict__ dst, int n) {
  int i = blockIdx.x * blockDim.x + threadIdx.x;
  if (i < n) dst[i] = (__bf16)src[i];
}

// src is [K, N] row-major fp32; dst is [N, K] row-major bf16 (i.e. W^T)
__global__ __launch_bounds__(256)
void cvt_transpose_bf16(const float* __restrict__ src,
                        __bf16* __restrict__ dst, int K, int N) {
  int i = blockIdx.x * blockDim.x + threadIdx.x;
  if (i < K * N) {
    int n = i / K, k = i % K;
    dst[i] = (__bf16)src[(size_t)k * N + n];
  }
}

// ------------------------------------------------- QKV GEMM + fused RoPE
// One wave per block. Wave tile: 16 rows (tokens) x 128 cols (one head's D).
// grid.x = 24 n-tiles (16 Q heads, 4 K heads, 4 V heads), grid.y = 256 m-tiles.
__global__ __launch_bounds__(32)
void qkv_rope_gemm(const __bf16* __restrict__ xb,
                   const __bf16* __restrict__ WqT, const __bf16* __restrict__ WkT,
                   const __bf16* __restrict__ WvT,
                   __bf16* __restrict__ Qr,  // [B,H,T,D]  (RoPE'd)
                   __bf16* __restrict__ Kr,  // [B,HKV,T,D] (RoPE'd)
                   __bf16* __restrict__ Vt)  // [B,HKV,D,T] (transposed)
{
  __shared__ __align__(16) __bf16 stage[16 * 128];  // 4KB output staging

  const int lane = threadIdx.x;
  const int col  = lane & 15;
  const int half = (lane >> 4) & 1;
  const int nt   = blockIdx.x;   // 0..23
  const int mt   = blockIdx.y;   // 0..255

  const __bf16* Wt; int mode, hidx;
  if (nt < 16)      { Wt = WqT + (size_t)nt * 128 * CC;        mode = 0; hidx = nt; }
  else if (nt < 20) { Wt = WkT + (size_t)(nt - 16) * 128 * CC; mode = 1; hidx = nt - 16; }
  else              { Wt = WvT + (size_t)(nt - 20) * 128 * CC; mode = 2; hidx = nt - 20; }

  v8f c[8];
#pragma unroll
  for (int ct = 0; ct < 8; ++ct) c[ct] = {};

  const __bf16* xrow = xb + (size_t)(mt * 16 + col) * CC;
  const __bf16* wrow = Wt + (size_t)col * CC;

  // Software-pipelined main loop: double-buffered fragments; sched-group
  // barriers pin (18 loads)(8 wmma)(18 loads)(8 wmma) per 64-wide k-step so
  // loads of the next half-step overlap the current WMMA burst.
  v16bf a0, a1, b0[8], b1[8];
  load_frags(xrow, wrow, 0, half, a0, b0);
  for (int k0 = 0; k0 < CC; k0 += 64) {
    load_frags(xrow, wrow, k0 + 32, half, a1, b1);
#pragma unroll
    for (int ct = 0; ct < 8; ++ct) c[ct] = wmma_bf16(a0, b0[ct], c[ct]);
    const int kn = (k0 + 64) & (CC - 1);   // wraps to 0 on last iter (benign reload)
    load_frags(xrow, wrow, kn, half, a0, b0);
#pragma unroll
    for (int ct = 0; ct < 8; ++ct) c[ct] = wmma_bf16(a1, b1[ct], c[ct]);
    SCHED_GROUP(0x020, 18, 0);
    SCHED_GROUP(0x008, 8, 0);
    SCHED_GROUP(0x020, 18, 0);
    SCHED_GROUP(0x008, 8, 0);
  }

  const int m0 = mt * 16;
  const int bq = m0 >> 11;          // batch (m-tiles never straddle batches)
  const int t0 = m0 & (TT - 1);

  if (mode == 2) {
    // V: stage [d][t] tile in LDS, then coalesced stores to Vt [B,HKV,D,T]
#pragma unroll
    for (int ct = 0; ct < 8; ++ct) {
      const int d = ct * 16 + col;
#pragma unroll
      for (int r = 0; r < 8; ++r) stage[d * 16 + r + 8 * half] = (__bf16)c[ct][r];
    }
    asm volatile("s_wait_dscnt 0" ::: "memory");
#pragma unroll
    for (int dd = 0; dd < 4; ++dd) {
      const int d = lane * 4 + dd;
      const __bf16* s = stage + d * 16;
      __bf16* g = Vt + (((size_t)bq * HKV + hidx) * DD + d) * TT + t0;
      *(v8bf*)g       = *(const v8bf*)s;
      *(v8bf*)(g + 8) = *(const v8bf*)(s + 8);
    }
  } else {
    __bf16* Out = (mode == 0) ? Qr : Kr;
    const int NH = (mode == 0) ? HH : HKV;
    // RoPE in-register (pairs d, d+64 live in the same lane: tiles ct, ct+4),
    // staged [t][d] in LDS.
#pragma unroll
    for (int ct = 0; ct < 4; ++ct) {
      const int i = ct * 16 + col;                 // d in [0,64)
      const float invf = __expf(-(float)i * (1.0f / 64.0f) * 9.210340371976184f); // 10000^(-i/64)
#pragma unroll
      for (int r = 0; r < 8; ++r) {
        const int tloc = r + 8 * half;
        float sn, cs;
        __sincosf((float)(t0 + tloc) * invf, &sn, &cs);
        const float x1 = c[ct][r], x2 = c[ct + 4][r];
        stage[tloc * 128 + i]      = (__bf16)(x1 * cs - x2 * sn);
        stage[tloc * 128 + 64 + i] = (__bf16)(x2 * cs + x1 * sn);
      }
    }
    asm volatile("s_wait_dscnt 0" ::: "memory");
    // Whole 16x128 tile is one contiguous 2048-element span in Out.
    __bf16* g = Out + (((size_t)bq * NH + hidx) * TT + t0) * DD;
#pragma unroll
    for (int j = 0; j < 8; ++j) {
      const int e = (j * 32 + lane) * 8;
      *(v8bf*)(g + e) = *(const v8bf*)(stage + e);
    }
  }
}

// ------------------------------------------------------- flash attention
// 1 wave per (b, h, 16-query tile). Online softmax, causal.
__global__ __launch_bounds__(32)
void flash_attn(const __bf16* __restrict__ Q,   // [B,H,T,D]
                const __bf16* __restrict__ K,   // [B,HKV,T,D]
                const __bf16* __restrict__ Vt,  // [B,HKV,D,T]
                __bf16* __restrict__ Aout)      // [B,T,H,D]
{
  __shared__ __align__(16) __bf16 pt[16 * 32];      // P tile transpose staging
  __shared__ __align__(16) __bf16 stage[16 * 128];  // output staging

  const int lane = threadIdx.x;
  const int col  = lane & 15;
  const int half = (lane >> 4) & 1;
  const int qt   = blockIdx.x;          // 0..127
  const int bh   = blockIdx.y;          // 0..31
  const int b = bh >> 4, h = bh & 15, hkv = h >> 2;

  const __bf16* Qb = Q  + (((size_t)b * HH  + h)   * TT + qt * 16) * DD;
  const __bf16* Kb = K  + (((size_t)b * HKV + hkv) * TT) * DD;
  const __bf16* Vb = Vt + (((size_t)b * HKV + hkv) * DD) * TT;

  // Q fragments: 16x128 held as 4 A-frags (resident for whole kernel)
  v16bf qf[4];
#pragma unroll
  for (int cch = 0; cch < 4; ++cch) {
    const __bf16* p = Qb + (size_t)col * DD + cch * 32 + half * 16;
    qf[cch] = pack_a(p, p + 8);
  }

  v8f o[8];
#pragma unroll
  for (int ct = 0; ct < 8; ++ct) o[ct] = {};
  float m_i[8], l_i[8];
#pragma unroll
  for (int r = 0; r < 8; ++r) { m_i[r] = -3.0e38f; l_i[r] = 0.0f; }

  const float scale = 0.08838834764831845f;  // 1/sqrt(128)
  const int kend = ((qt + 1) * 16 + 31) >> 5;

  for (int kb = 0; kb < kend; ++kb) {
    const int key0 = kb * 32;

    // --- all K fragments for this 32-key block, then 8 S-WMMAs ---
    v16bf kf[8];  // [ns][chunk]
#pragma unroll
    for (int ns = 0; ns < 2; ++ns) {
      const int key = key0 + ns * 16 + col;
#pragma unroll
      for (int cch = 0; cch < 4; ++cch) {
        const __bf16* p = Kb + (size_t)key * DD + cch * 32 + half * 16;
        kf[ns * 4 + cch] = *(const v16bf*)p;
      }
    }
    v8f s[2];
#pragma unroll
    for (int ns = 0; ns < 2; ++ns) {
      s[ns] = {};
#pragma unroll
      for (int cch = 0; cch < 4; ++cch)
        s[ns] = wmma_bf16(qf[cch], kf[ns * 4 + cch], s[ns]);
    }

    // --- issue V loads NOW; softmax VALU below hides their latency ---
    v16bf vf[8];
#pragma unroll
    for (int ct = 0; ct < 8; ++ct) {
      const int d = ct * 16 + col;
      const __bf16* vp = Vb + (size_t)d * TT + key0 + half * 16;
      vf[ct] = *(const v16bf*)vp;
    }

    // --- scale + causal mask + online softmax (register/shuffle only) ---
#pragma unroll
    for (int r = 0; r < 8; ++r) {
      const int q = qt * 16 + r + 8 * half;
      {
        const int k0c = key0 + col, k1c = key0 + 16 + col;
        float v0 = s[0][r] * scale, v1 = s[1][r] * scale;
        s[0][r] = (k0c <= q) ? v0 : -3.0e38f;
        s[1][r] = (k1c <= q) ? v1 : -3.0e38f;
      }
      float mx = fmaxf(s[0][r], s[1][r]);
      mx = fmaxf(mx, __shfl_xor(mx, 1, 32));
      mx = fmaxf(mx, __shfl_xor(mx, 2, 32));
      mx = fmaxf(mx, __shfl_xor(mx, 4, 32));
      mx = fmaxf(mx, __shfl_xor(mx, 8, 32));
      const float mnew  = fmaxf(m_i[r], mx);
      const float alpha = __expf(m_i[r] - mnew);
      const float e0 = __expf(s[0][r] - mnew);
      const float e1 = __expf(s[1][r] - mnew);
      float rs = e0 + e1;
      rs += __shfl_xor(rs, 1, 32);
      rs += __shfl_xor(rs, 2, 32);
      rs += __shfl_xor(rs, 4, 32);
      rs += __shfl_xor(rs, 8, 32);
      l_i[r] = l_i[r] * alpha + rs;
      m_i[r] = mnew;
#pragma unroll
      for (int ct = 0; ct < 8; ++ct) o[ct][r] *= alpha;
      s[0][r] = e0; s[1][r] = e1;
    }

    // --- P: C-layout -> LDS -> A-layout ---
#pragma unroll
    for (int r = 0; r < 8; ++r) {
      pt[(r + 8 * half) * 32 + col]      = (__bf16)s[0][r];
      pt[(r + 8 * half) * 32 + 16 + col] = (__bf16)s[1][r];
    }
    asm volatile("s_wait_dscnt 0" ::: "memory");  // DS in-order per wave; be explicit
    const __bf16* pp = pt + col * 32 + half * 16;
    v16bf pf = pack_a(pp, pp + 8);

    // --- O += P V ---
#pragma unroll
    for (int ct = 0; ct < 8; ++ct) o[ct] = wmma_bf16(pf, vf[ct], o[ct]);

    SCHED_GROUP(0x020, 16, 0);  // K frag loads
    SCHED_GROUP(0x008, 8, 0);   // S wmmas
    SCHED_GROUP(0x020, 16, 0);  // V frag loads (before softmax VALU)
    SCHED_GROUP(0x008, 8, 0);   // O wmmas
  }

  // normalize, stage in LDS, coalesced store [B,T,H,D]
#pragma unroll
  for (int r = 0; r < 8; ++r) l_i[r] = 1.0f / l_i[r];
#pragma unroll
  for (int ct = 0; ct < 8; ++ct) {
    const int d = ct * 16 + col;
#pragma unroll
    for (int r = 0; r < 8; ++r)
      stage[(r + 8 * half) * 128 + d] = (__bf16)(o[ct][r] * l_i[r]);
  }
  asm volatile("s_wait_dscnt 0" ::: "memory");
  const int t0 = qt * 16;
  __bf16* g = Aout + (((size_t)b * TT + t0) * HH + h) * DD;
#pragma unroll
  for (int row = 0; row < 16; ++row) {
    const int e = lane * 4;
    *(v4bf*)(g + (size_t)row * (HH * DD) + e) = *(const v4bf*)(stage + row * 128 + e);
  }
}

// ------------------------------------------------------- output projection
// grid.x = 16 n-tiles(128), grid.y = 256 m-tiles; fp32 store.
__global__ __launch_bounds__(32)
void oproj_gemm(const __bf16* __restrict__ Ab,   // [4096, 2048] bf16
                const __bf16* __restrict__ WoT,  // [2048, 2048] bf16 (Wo^T)
                float* __restrict__ out)         // [4096, 2048] fp32
{
  __shared__ __align__(16) float stagef[16 * 128];  // 8KB output staging

  const int lane = threadIdx.x;
  const int col  = lane & 15;
  const int half = (lane >> 4) & 1;
  const int nt   = blockIdx.x;
  const int mt   = blockIdx.y;

  v8f c[8];
#pragma unroll
  for (int ct = 0; ct < 8; ++ct) c[ct] = {};

  const __bf16* arow = Ab + (size_t)(mt * 16 + col) * CC;
  const __bf16* wrow = WoT + (size_t)(nt * 128 + col) * CC;

  v16bf a0, a1, b0[8], b1[8];
  load_frags(arow, wrow, 0, half, a0, b0);
  for (int k0 = 0; k0 < CC; k0 += 64) {
    load_frags(arow, wrow, k0 + 32, half, a1, b1);
#pragma unroll
    for (int ct = 0; ct < 8; ++ct) c[ct] = wmma_bf16(a0, b0[ct], c[ct]);
    const int kn = (k0 + 64) & (CC - 1);
    load_frags(arow, wrow, kn, half, a0, b0);
#pragma unroll
    for (int ct = 0; ct < 8; ++ct) c[ct] = wmma_bf16(a1, b1[ct], c[ct]);
    SCHED_GROUP(0x020, 18, 0);
    SCHED_GROUP(0x008, 8, 0);
    SCHED_GROUP(0x020, 18, 0);
    SCHED_GROUP(0x008, 8, 0);
  }

#pragma unroll
  for (int ct = 0; ct < 8; ++ct) {
#pragma unroll
    for (int r = 0; r < 8; ++r)
      stagef[(r + 8 * half) * 128 + ct * 16 + col] = c[ct][r];
  }
  asm volatile("s_wait_dscnt 0" ::: "memory");
  const int m0 = mt * 16;
  float* g = out + (size_t)m0 * CC + nt * 128;
#pragma unroll
  for (int row = 0; row < 16; ++row) {
    const int e = lane * 4;
    *(v4f*)(g + (size_t)row * CC + e) = *(const v4f*)(stagef + row * 128 + e);
  }
}

// ---------------------------------------------------------------- launch
extern "C" void kernel_launch(void* const* d_in, const int* in_sizes, int n_in,
                              void* d_out, int out_size, void* d_ws, size_t ws_size,
                              hipStream_t stream) {
  (void)in_sizes; (void)n_in; (void)out_size; (void)ws_size;
  const float* x  = (const float*)d_in[0];
  const float* Wq = (const float*)d_in[1];
  const float* Wk = (const float*)d_in[2];
  const float* Wv = (const float*)d_in[3];
  const float* Wo = (const float*)d_in[4];
  // d_in[5] (mask) unused: causal mask applied analytically.

  char* ws = (char*)d_ws;
  __bf16* xb   = (__bf16*)(ws);                         // 16 MB  [4096,2048]
  __bf16* WqT  = (__bf16*)(ws + 16777216);              //  8 MB  [2048,2048]
  __bf16* WkT  = (__bf16*)(ws + 25165824);              //  2 MB  [512,2048]
  __bf16* WvT  = (__bf16*)(ws + 27262976);              //  2 MB  [512,2048]
  __bf16* WoT  = (__bf16*)(ws + 29360128);              //  8 MB  [2048,2048]
  __bf16* Qr   = (__bf16*)(ws + 37748736);              // 16 MB  [B,H,T,D]
  __bf16* Kr   = (__bf16*)(ws + 54525952);              //  4 MB  [B,HKV,T,D]
  __bf16* Vt   = (__bf16*)(ws + 58720256);              //  4 MB  [B,HKV,D,T]
  __bf16* Aout = (__bf16*)(ws + 62914560);              // 16 MB  [B,T,H,D]
  float*  out  = (float*)d_out;

  // 1) converts / transposes
  {
    int n = BB * TT * CC;  // 8388608
    cvt_f32_bf16<<<(n + 255) / 256, 256, 0, stream>>>(x, xb, n);
  }
  cvt_transpose_bf16<<<(CC * HH * DD + 255) / 256, 256, 0, stream>>>(Wq, WqT, CC, HH * DD);
  cvt_transpose_bf16<<<(CC * HKV * DD + 255) / 256, 256, 0, stream>>>(Wk, WkT, CC, HKV * DD);
  cvt_transpose_bf16<<<(CC * HKV * DD + 255) / 256, 256, 0, stream>>>(Wv, WvT, CC, HKV * DD);
  cvt_transpose_bf16<<<(HH * DD * CC + 255) / 256, 256, 0, stream>>>(Wo, WoT, HH * DD, CC);

  // 2) fused QKV projection + RoPE (+ V transpose)
  qkv_rope_gemm<<<dim3(24, 256), 32, 0, stream>>>(xb, WqT, WkT, WvT, Qr, Kr, Vt);

  // 3) flash attention
  flash_attn<<<dim3(TT / 16, BB * HH), 32, 0, stream>>>(Qr, Kr, Vt, Aout);

  // 4) output projection (fp32 out)
  oproj_gemm<<<dim3(16, 256), 32, 0, stream>>>(Aout, WoT, out);
}
